// MultiheadAttention_34471407518030
// MI455X (gfx1250) — matrix operand order repaired and checked by reading, not compiled, not run on hardware
//
#include <hip/hip_runtime.h>
#include <hip/hip_bf16.h>

// ---------------------------------------------------------------------------
// MI455X (gfx1250) GQA attention with RoPE, bf16 WMMA + fp32 accumulate.
// Compute-bound (~165 GFLOP vs ~150MB traffic): all heavy math goes through
// v_wmma_f32_16x16x32_bf16; staging uses global_load_async_to_lds_b128 with
// ping-pong LDS double buffering; softmax row reductions use DPP16
// butterflies (no LDS round trips).
// ---------------------------------------------------------------------------

typedef __bf16 bf16_t;
typedef __attribute__((ext_vector_type(16))) __bf16 v16bf;
typedef __attribute__((ext_vector_type(8)))  __bf16 v8bf;
typedef __attribute__((ext_vector_type(8)))  float  v8f;

#define S_LEN 2048
#define DIN   2048
#define NH    16
#define NKV   4
#define HD    128
#define DMODEL 2048   // NH*HD == D_OUT == D_IN

// ----------------------- async global -> LDS copy --------------------------
// GLOBAL_LOAD_ASYNC_TO_LDS_B128: per-lane 16B memory->LDS, tracked by ASYNCcnt.
// Flat addresses of __shared__ objects carry the LDS byte offset in [31:0].
__device__ __forceinline__ void async_copy16(bf16_t* lds_dst,
                                             const bf16_t* gsrc) {
  unsigned int lds = (unsigned int)(unsigned long long)lds_dst;
  asm volatile("global_load_async_to_lds_b128 %0, %1, off"
               :: "v"(lds), "v"(gsrc)
               : "memory");
}
__device__ __forceinline__ void wait_async0() {
  asm volatile("s_wait_asynccnt 0x0" ::: "memory");
}

// ------------------------- WMMA fragment helpers ---------------------------
// A fragment 16x32 bf16 (ISA 7.12.2): lane L holds row M=L&15.
//   lanes 0-15 : K = 0..7  (VGPR0-3), K = 16..23 (VGPR4-7)
//   lanes 16-31: K = 8..15 (VGPR0-3), K = 24..31 (VGPR4-7)
__device__ __forceinline__ v16bf load_a_frag(const bf16_t* base, int ld) {
  const int lane = threadIdx.x & 31;
  const int m = lane & 15;
  const int g = (lane & 16) ? 8 : 0;
  const bf16_t* p = base + (size_t)m * ld + g;
  v8bf lo = *(const v8bf*)(p);        // K = g .. g+7
  v8bf hi = *(const v8bf*)(p + 16);   // K = 16+g .. 16+g+7
  return __builtin_shufflevector(lo, hi, 0,1,2,3,4,5,6,7,8,9,10,11,12,13,14,15);
}

// B fragment 32x16 bf16: lane L holds column N=L&15; lanes 0-15 carry K=0..7,16..23,
// lanes 16-31 carry K=8..15,24..31. Source stored as [N][K] row-major, which
// matches W[out,in] weights directly.
__device__ __forceinline__ v16bf load_b_frag(const bf16_t* base, int ld) {
  const int lane = threadIdx.x & 31;
  const int n = lane & 15;
  const int kb = (lane & 16) ? 8 : 0;
  const bf16_t* p = base + (size_t)n * ld + kb;
  v8bf lo = *(const v8bf*)(p);
  v8bf hi = *(const v8bf*)(p + 16);
  return __builtin_shufflevector(lo, hi, 0,1,2,3,4,5,6,7,8,9,10,11,12,13,14,15);
}

__device__ __forceinline__ v8f wmma_bf16(v16bf a, v16bf b, v8f c) {
  return __builtin_amdgcn_wmma_f32_16x16x32_bf16(
      false, a, false, b, (short)0, c, false, false);
}

__device__ __forceinline__ v8f vzero8f() {
  v8f z = {0.f,0.f,0.f,0.f,0.f,0.f,0.f,0.f};
  return z;
}

// --------------------- DPP16 butterfly row reductions ----------------------
// Full reduction over each 16-lane group (one DPP row on wave32):
//   xor1: quad_perm(1,0,3,2)=0xB1, xor2: quad_perm(2,3,0,1)=0x4E,
//   i^7 : row_half_mirror=0x141 (merges the two quads of a half-row),
//   i^15: row_mirror=0x140 (merges the two half-rows).
// Never crosses the 16-lane boundary, so the two C-matrix row groups
// (lanes 0-15 / 16-31) stay independent. Pure VALU, no LDS, no waits.
template <int CTRL>
__device__ __forceinline__ float dpp_f(float v) {
  return __int_as_float(__builtin_amdgcn_update_dpp(
      0, __float_as_int(v), CTRL, 0xf, 0xf, true));
}
__device__ __forceinline__ float rmax16(float v) {
  v = fmaxf(v, dpp_f<0xB1>(v));
  v = fmaxf(v, dpp_f<0x4E>(v));
  v = fmaxf(v, dpp_f<0x141>(v));
  v = fmaxf(v, dpp_f<0x140>(v));
  return v;
}
__device__ __forceinline__ float rsum16(float v) {
  v += dpp_f<0xB1>(v);
  v += dpp_f<0x4E>(v);
  v += dpp_f<0x141>(v);
  v += dpp_f<0x140>(v);
  return v;
}

// ------------------------------ cast kernels -------------------------------
__global__ void cast_f32_bf16(const float* __restrict__ in,
                              bf16_t* __restrict__ out, int n) {
  int i = blockIdx.x * blockDim.x + threadIdx.x;
  if (i < n) out[i] = (bf16_t)in[i];
}

// RoPE (GPT-NeoX style) + permute [S, heads*HD] fp32 -> [heads][S][HD] bf16.
// scale folded into q (rope is linear, so scale*rope(q) == rope(scale*q)).
__global__ void rope_permute(const float* __restrict__ in,
                             bf16_t* __restrict__ out,
                             int heads, float scale) {
  int idx = blockIdx.x * blockDim.x + threadIdx.x;
  int lim = S_LEN * heads * 64;
  if (idx >= lim) return;
  int dh = idx & 63;
  int h  = (idx >> 6) % heads;
  int s  = idx / (64 * heads);
  const float* row = in + (size_t)s * heads * HD + (size_t)h * HD;
  float x1 = row[dh];
  float x2 = row[dh + 64];
  // inv_freq = theta^(-dh/64); ln(10000)/64 = 0.14391156831
  float w = __expf(-(float)dh * 0.14391156831f);
  float ang = (float)s * w;
  float sn, cs;
  __sincosf(ang, &sn, &cs);
  bf16_t* orow = out + ((size_t)h * S_LEN + s) * HD;
  orow[dh]      = (bf16_t)((x1 * cs - x2 * sn) * scale);
  orow[dh + 64] = (bf16_t)((x2 * cs + x1 * sn) * scale);
}

// [S, NKV*HD] fp32 -> [NKV][S][HD] bf16 (values, no rotation)
__global__ void permute_cast_v(const float* __restrict__ in,
                               bf16_t* __restrict__ out) {
  int idx = blockIdx.x * blockDim.x + threadIdx.x;
  if (idx >= S_LEN * NKV * HD) return;
  int d = idx & 127;
  int h = (idx >> 7) & 3;
  int s = idx >> 9;
  out[((size_t)h * S_LEN + s) * HD + d] =
      (bf16_t)in[(size_t)s * (NKV * HD) + (size_t)h * HD + d];
}

// ------------------------------ bf16 GEMM ----------------------------------
// C[M,N] fp32 = A[M,K] bf16 * B[N,K]^T bf16. N,K compile-time so the epilogue
// folds into immediate-offset stores. 128x128 block, 8 waves of 64x32,
// K-step 32, async double-buffered LDS staging.
template <int N, int K>
__global__ __launch_bounds__(256)
void gemm_bf16_nt(const bf16_t* __restrict__ A, const bf16_t* __restrict__ B,
                  float* __restrict__ C) {
  __shared__ __align__(16) bf16_t As[2][128 * 32];
  __shared__ __align__(16) bf16_t Bs[2][128 * 32];
  const int t    = threadIdx.x;
  const int wv   = t >> 5;
  const int wm   = wv >> 2;  // 0..1
  const int wn   = wv & 3;   // 0..3
  const int rb   = blockIdx.y * 128;
  const int cb   = blockIdx.x * 128;
  const int lane = t & 31;
  const int n    = lane & 15;
  const int half = lane >> 4;

  v8f acc[4][2];
#pragma unroll
  for (int i = 0; i < 4; ++i)
#pragma unroll
    for (int j = 0; j < 2; ++j) acc[i][j] = vzero8f();

  // Each thread stages 2x16B of A and of B per tile (512 vectors each).
  const int r0 = t >> 2;          // 0..63
  const int c8 = (t & 3) * 8;

  auto stage = [&](int k0, int buf) {
#pragma unroll
    for (int i = 0; i < 2; ++i) {
      int r = r0 + i * 64;
      async_copy16(&As[buf][r * 32 + c8], &A[(size_t)(rb + r) * K + k0 + c8]);
      async_copy16(&Bs[buf][r * 32 + c8], &B[(size_t)(cb + r) * K + k0 + c8]);
    }
  };

  stage(0, 0);
  for (int k0 = 0; k0 < K; k0 += 32) {
    const int buf = (k0 >> 5) & 1;
    wait_async0();        // my async writes into `buf` (and older) done
    __syncthreads();      // everyone's staging done / previous reads done
    if (k0 + 32 < K) stage(k0 + 32, buf ^ 1);   // prefetch next tile

    v16bf a[4], b[2];
#pragma unroll
    for (int i = 0; i < 4; ++i)
      a[i] = load_a_frag(&As[buf][(wm * 64 + i * 16) * 32], 32);
#pragma unroll
    for (int j = 0; j < 2; ++j)
      b[j] = load_b_frag(&Bs[buf][(wn * 32 + j * 16) * 32], 32);
#pragma unroll
    for (int i = 0; i < 4; ++i)
#pragma unroll
      for (int j = 0; j < 2; ++j) acc[i][j] = wmma_bf16(a[i], b[j], acc[i][j]);
  }

  // Epilogue: one base address per fragment, 8 immediate-offset stores.
#pragma unroll
  for (int i = 0; i < 4; ++i)
#pragma unroll
    for (int j = 0; j < 2; ++j) {
      float* p = C + (size_t)(rb + wm * 64 + i * 16 + 8 * half) * N +
                 (cb + wn * 32 + j * 16 + n);
#pragma unroll
      for (int r = 0; r < 8; ++r) p[(size_t)r * N] = acc[i][j][r];
    }
}

// --------------------------- flash attention -------------------------------
// Block = 4 waves = 64 q-rows of one query head. Loops over the 4 kv heads,
// online softmax over 32-key tiles (causal), averages 1/NKV over kv heads.
__global__ __launch_bounds__(128)
void attn_kernel(const bf16_t* __restrict__ qbf,   // [NH][S][HD] roped+scaled
                 const bf16_t* __restrict__ kbf,   // [NKV][S][HD] roped
                 const bf16_t* __restrict__ vbf,   // [NKV][S][HD]
                 bf16_t* __restrict__ attn) {      // [S][NH*HD]
  __shared__ __align__(16) bf16_t Kt[32 * 128];    // [key][d]  (= B layout)
  __shared__ __align__(16) bf16_t Vt[128 * 48];    // [d][key], stride 48
  __shared__ __align__(16) bf16_t Pbuf[4][16 * 32];

  const int t        = threadIdx.x;
  const int wv       = t >> 5;
  const int lane     = t & 31;
  const int n        = lane & 15;
  const int half     = lane >> 4;
  const int h        = blockIdx.y;
  const int q0       = blockIdx.x * 64;
  const int row_base = q0 + wv * 16;

  // Q fragments for this wave's 16 rows (K = HD = 128 -> 4 frags of 32).
  const bf16_t* qhead = qbf + ((size_t)h * S_LEN + row_base) * HD;
  v16bf qa[4];
#pragma unroll
  for (int kk = 0; kk < 4; ++kk) qa[kk] = load_a_frag(qhead + kk * 32, HD);

  v8f ofin[8];
#pragma unroll
  for (int f = 0; f < 8; ++f) ofin[f] = vzero8f();

  const int jmax = (q0 + 64 + 31) / 32;

  for (int g = 0; g < NKV; ++g) {
    const bf16_t* kg = kbf + (size_t)g * S_LEN * HD;
    const bf16_t* vg = vbf + (size_t)g * S_LEN * HD;

    v8f o[8];
#pragma unroll
    for (int f = 0; f < 8; ++f) o[f] = vzero8f();
    float mrun[8], lrun[8];
#pragma unroll
    for (int i = 0; i < 8; ++i) { mrun[i] = -3.0e38f; lrun[i] = 0.f; }

    for (int j = 0; j < jmax; ++j) {
      const int jb = j * 32;
      __syncthreads();  // previous tile fully consumed
      // K tile [32][128] via async copies: 256 x 16B, 2 per thread.
#pragma unroll
      for (int i = 0; i < 2; ++i) {
        int vi  = t + i * 128;          // 0..255
        int key = vi >> 3;              // 8 copies per 256B key row
        int d8  = (vi & 7) * 16;
        async_copy16(&Kt[key * 128 + d8],
                     &kg[(size_t)(jb + key) * HD + d8]);
      }
      // V transposed (scalar, it's a transpose): Vt[d][key].
#pragma unroll
      for (int i = 0; i < 32; ++i) {
        int e   = t + i * 128;
        int key = e >> 7;
        int d   = e & 127;
        Vt[d * 48 + key] = vg[(size_t)(jb + key) * HD + d];
      }
      wait_async0();
      __syncthreads();

      if (jb <= row_base + 15) {        // uniform per wave (EXEC all-1)
        // S = Q * K^T : two 16-col fragments covering 32 keys.
        v8f c0 = vzero8f(), c1 = vzero8f();
#pragma unroll
        for (int kk = 0; kk < 4; ++kk) {
          c0 = wmma_bf16(qa[kk], load_b_frag(&Kt[0 * 128 + kk * 32], 128), c0);
          c1 = wmma_bf16(qa[kk], load_b_frag(&Kt[16 * 128 + kk * 32], 128), c1);
        }
        // Causal mask (C layout: row = i + 8*half, col = jb + nt*16 + n).
        if (jb + 31 > row_base) {
#pragma unroll
          for (int i = 0; i < 8; ++i) {
            int row = row_base + i + 8 * half;
            if (jb + n > row)      c0[i] = -3.0e38f;
            if (jb + 16 + n > row) c1[i] = -3.0e38f;
          }
        }
        // Online softmax update (DPP16 butterflies, no LDS).
        float alpha[8];
#pragma unroll
        for (int i = 0; i < 8; ++i) {
          float tm = rmax16(fmaxf(c0[i], c1[i]));
          float mn = fmaxf(mrun[i], tm);
          alpha[i] = __expf(mrun[i] - mn);
          mrun[i] = mn;
        }
#pragma unroll
        for (int i = 0; i < 8; ++i) {
          float p0 = __expf(c0[i] - mrun[i]);
          float p1 = __expf(c1[i] - mrun[i]);
          c0[i] = p0; c1[i] = p1;
          lrun[i] = lrun[i] * alpha[i] + rsum16(p0 + p1);
        }
        // C-layout P -> A-layout via per-wave LDS round trip.
        bf16_t* pb = Pbuf[wv];
#pragma unroll
        for (int i = 0; i < 8; ++i) {
          int m = i + 8 * half;
          pb[m * 32 + n]      = (bf16_t)c0[i];
          pb[m * 32 + 16 + n] = (bf16_t)c1[i];
        }
        // Rescale running O, then O += P * V (K=32 per WMMA).
#pragma unroll
        for (int f = 0; f < 8; ++f)
#pragma unroll
          for (int i = 0; i < 8; ++i) o[f][i] *= alpha[i];
        v16bf pa = load_a_frag(pb, 32);
#pragma unroll
        for (int f = 0; f < 8; ++f)
          o[f] = wmma_bf16(pa, load_b_frag(&Vt[(f * 16) * 48], 48), o[f]);
      }
    }
    // ofin += O_g / (NKV * l_g)
#pragma unroll
    for (int i = 0; i < 8; ++i) {
      float inv = 0.25f / lrun[i];
#pragma unroll
      for (int f = 0; f < 8; ++f) ofin[f][i] += o[f][i] * inv;
    }
  }

  // Write attn rows as bf16 [S][NH*HD] for the output projection GEMM.
#pragma unroll
  for (int f = 0; f < 8; ++f)
#pragma unroll
    for (int i = 0; i < 8; ++i) {
      int row = row_base + i + 8 * half;
      attn[(size_t)row * DMODEL + (size_t)h * HD + f * 16 + n] =
          (bf16_t)ofin[f][i];
    }
}

// ------------------------------- launcher ----------------------------------
extern "C" void kernel_launch(void* const* d_in, const int* in_sizes, int n_in,
                              void* d_out, int out_size, void* d_ws,
                              size_t ws_size, hipStream_t stream) {
  (void)in_sizes; (void)n_in; (void)out_size; (void)ws_size;
  const float* x  = (const float*)d_in[0];
  const float* Wq = (const float*)d_in[1];
  const float* Wk = (const float*)d_in[2];
  const float* Wv = (const float*)d_in[3];
  const float* Wo = (const float*)d_in[4];
  float* out = (float*)d_out;

  char* ws = (char*)d_ws;
  size_t off = 0;
  auto alloc = [&](size_t bytes) -> void* {
    void* p = ws + off;
    off = (off + bytes + 255) & ~(size_t)255;
    return p;
  };

  bf16_t* xb   = (bf16_t*)alloc((size_t)S_LEN * DIN * 2);
  bf16_t* Wqb  = (bf16_t*)alloc((size_t)DMODEL * DIN * 2);
  bf16_t* Wkb  = (bf16_t*)alloc((size_t)NKV * HD * DIN * 2);
  bf16_t* Wvb  = (bf16_t*)alloc((size_t)NKV * HD * DIN * 2);
  bf16_t* Wob  = (bf16_t*)alloc((size_t)DMODEL * DMODEL * 2);
  float*  qf   = (float*) alloc((size_t)S_LEN * DMODEL * 4);
  float*  kf   = (float*) alloc((size_t)S_LEN * NKV * HD * 4);
  float*  vf   = (float*) alloc((size_t)S_LEN * NKV * HD * 4);
  bf16_t* qb   = (bf16_t*)alloc((size_t)NH  * S_LEN * HD * 2);
  bf16_t* kb   = (bf16_t*)alloc((size_t)NKV * S_LEN * HD * 2);
  bf16_t* vb   = (bf16_t*)alloc((size_t)NKV * S_LEN * HD * 2);
  bf16_t* attn = (bf16_t*)alloc((size_t)S_LEN * DMODEL * 2);

  const int T = 256;
  auto blocks = [](int nelem, int th) { return (nelem + th - 1) / th; };

  // 1) casts to bf16
  cast_f32_bf16<<<blocks(S_LEN * DIN, T), T, 0, stream>>>(x, xb, S_LEN * DIN);
  cast_f32_bf16<<<blocks(DMODEL * DIN, T), T, 0, stream>>>(Wq, Wqb, DMODEL * DIN);
  cast_f32_bf16<<<blocks(NKV * HD * DIN, T), T, 0, stream>>>(Wk, Wkb, NKV * HD * DIN);
  cast_f32_bf16<<<blocks(NKV * HD * DIN, T), T, 0, stream>>>(Wv, Wvb, NKV * HD * DIN);
  cast_f32_bf16<<<blocks(DMODEL * DMODEL, T), T, 0, stream>>>(Wo, Wob, DMODEL * DMODEL);

  // 2) projections: y = x @ W^T  (W stored [N,K] row-major = B layout)
  gemm_bf16_nt<DMODEL, DIN><<<dim3(DMODEL / 128, S_LEN / 128), 256, 0, stream>>>(
      xb, Wqb, qf);
  gemm_bf16_nt<NKV * HD, DIN><<<dim3((NKV * HD) / 128, S_LEN / 128), 256, 0, stream>>>(
      xb, Wkb, kf);
  gemm_bf16_nt<NKV * HD, DIN><<<dim3((NKV * HD) / 128, S_LEN / 128), 256, 0, stream>>>(
      xb, Wvb, vf);

  // 3) RoPE (+1/sqrt(HD) folded into q) and head-major bf16 repack
  const float qscale = 0.088388347648f; // 1/sqrt(128)
  rope_permute<<<blocks(S_LEN * NH * 64, T), T, 0, stream>>>(qf, qb, NH, qscale);
  rope_permute<<<blocks(S_LEN * NKV * 64, T), T, 0, stream>>>(kf, kb, NKV, 1.0f);
  permute_cast_v<<<blocks(S_LEN * NKV * HD, T), T, 0, stream>>>(vf, vb);

  // 4) causal GQA flash attention (mean over kv heads)
  attn_kernel<<<dim3(S_LEN / 64, NH), 128, 0, stream>>>(qb, kb, vb, attn);

  // 5) output projection -> fp32 d_out
  gemm_bf16_nt<DMODEL, DMODEL><<<dim3(DMODEL / 128, S_LEN / 128), 256, 0, stream>>>(
      attn, Wob, out);
}
